// BidirectionalLSTM_29987461660981
// MI455X (gfx1250) — matrix operand order repaired and checked by reading, not compiled, not run on hardware
//
#include <hip/hip_runtime.h>

typedef __attribute__((ext_vector_type(16))) _Float16 v16h;
typedef __attribute__((ext_vector_type(8)))  _Float16 v8h;
typedef __attribute__((ext_vector_type(8)))  float    v8f;

namespace {
constexpr int kB = 32, kT = 2048, kD = 128, kH = 256, kG = 1024;
constexpr int kNTiles    = kG / 16;   // 64 N-tiles of the gate dimension
constexpr int kKTilesIH  = kD / 32;   // 4 K-slices for the input projection
constexpr int kKTilesHH  = kH / 32;   // 8 K-slices for the recurrent matmul
constexpr size_t kFragHalves = 32 * 16; // one B-fragment: 32 lanes x 16 halves = 1KB
}

__device__ __forceinline__ float dev_sigmoid(float x) {
  return 1.0f / (1.0f + __expf(-x));
}
__device__ __forceinline__ float dev_tanh(float x) {
  x = fminf(fmaxf(x, -15.0f), 15.0f);
  float e = __expf(-2.0f * x);
  return (1.0f - e) / (1.0f + e);
}

// Issue one per-lane 16B async copy global -> LDS (uses ASYNCcnt, not LOADcnt).
// LDS address = low 32 bits of the generic pointer (flat->LDS aperture mapping).
__device__ __forceinline__ void async_copy_b128(const float* gsrc, float* ldst) {
  unsigned       lo = (unsigned)(size_t)(void*)ldst;
  unsigned long long ga = (unsigned long long)(size_t)(const void*)gsrc;
  asm volatile("global_load_async_to_lds_b128 %0, %1, off"
               :: "v"(lo), "v"(ga) : "memory");
}
__device__ __forceinline__ void wait_async0() {
  asm volatile("s_wait_asynccnt 0x0" ::: "memory");
}

// Pre-swizzle an f32 weight matrix [K][1024] (row-major) into f16 WMMA
// B-fragments: fragment(nTile,kTile) -> 32 lanes x 16 halves contiguous.
// Lane layout (16-bit B 32x16): N = nTile*16 + lane%16,
//                               K = kTile*32 + (lane/16)*8 + (j%8) + (j/8)*16.
__global__ void k_swizzle(const float* __restrict__ W,
                          _Float16* __restrict__ dst, int kTiles) {
  int idx = blockIdx.x * 256 + threadIdx.x;
  int total = kNTiles * kTiles * 512;
  if (idx >= total) return;
  int j    = idx & 15;
  int lane = (idx >> 4) & 31;
  int fk   = (idx >> 9) % kTiles;
  int nT   = (idx >> 9) / kTiles;
  int n = nT * 16 + (lane & 15);
  int K = fk * 32 + ((lane >> 4) << 3) + (j & 7) + ((j >> 3) << 4);
  dst[idx] = (_Float16)W[K * kG + n];
}

// Input projection for both directions:
//   xg[dir][t*32+b][g] = x'[b][t][:] . W_ih[dir][:, g] + bias[dir][g]
// where x' is time-flipped for dir==1. One 16x16 C-tile per (wave, n),
// each wave covers 16 rows x 64 gate-columns, K=128 via 4 wmma k-steps.
__global__ __launch_bounds__(256) void k_xproj(
    const float* __restrict__ x, const _Float16* __restrict__ wihSwz,
    const float* __restrict__ bf, const float* __restrict__ bb,
    float* __restrict__ xg) {
  const int wave = threadIdx.x >> 5, lane = threadIdx.x & 31;
  const int mloc = lane & 15, half = lane >> 4;
  constexpr int nBlocks = kNTiles / 4;          // 16
  constexpr int mBlocks = (kT * kB) / 16 / 8;   // 512
  int wg = blockIdx.x;
  int dir = wg / (mBlocks * nBlocks);
  wg %= mBlocks * nBlocks;
  int mBlk = wg / nBlocks, nBlk = wg % nBlocks;
  int mTile = mBlk * 8 + wave;
  int row0 = mTile * 16;
  const float* bias = dir ? bb : bf;
  const _Float16* wd =
      wihSwz + (size_t)dir * kNTiles * kKTilesIH * kFragHalves;
  float* xgd = xg + (size_t)dir * kT * kB * kG;

  int row = row0 + mloc;            // row = t*32 + b
  int t = row >> 5, b = row & 31;
  int ts = dir ? (kT - 1 - t) : t;  // flip time for backward direction
  const float* arow = x + ((size_t)b * kT + ts) * kD;

  v8f acc[4] = {};
#pragma unroll
  for (int k = 0; k < kKTilesIH; ++k) {
    // A-fragment (16-bit A 16x32): lane holds K = k*32 + {8*half+0..7, 16+8*half+0..7}
    const float* p = arow + k * 32 + 8 * half;
    union { v16h v; _Float16 e[16]; } a;
#pragma unroll
    for (int j = 0; j < 8; ++j) a.e[j] = (_Float16)p[j];
#pragma unroll
    for (int j = 0; j < 8; ++j) a.e[8 + j] = (_Float16)p[16 + j];
#pragma unroll
    for (int n = 0; n < 4; ++n) {
      int nT = nBlk * 4 + n;
      v16h bfr = *(const v16h*)(wd + (size_t)(nT * kKTilesIH + k) * kFragHalves +
                                lane * 16);
      acc[n] = __builtin_amdgcn_wmma_f32_16x16x32_f16(
          false, a.v, false, bfr, (short)0, acc[n], false, false);
    }
  }
#pragma unroll
  for (int n = 0; n < 4; ++n) {
    int col = nBlk * 64 + n * 16 + mloc;
    float bv = bias[col];
#pragma unroll
    for (int v = 0; v < 8; ++v) {
      int r = row0 + v + 8 * half;  // C layout: row = v + 8*(lane/16)
      __builtin_nontemporal_store(acc[n][v] + bv, &xgd[(size_t)r * kG + col]);
    }
  }
}

// Recurrence. grid = 4 workgroups: (dir, batch-half). 1024 threads = 32 waves.
// W_hh stays resident in VGPRs (16 v16h fragments/wave = 128 VGPRs) for all
// 2048 steps; h (f16) and c (f32) live in LDS; gates exchanged via LDS.
// xg[t+1] is staged into a double-buffered LDS region with
// GLOBAL_LOAD_ASYNC_TO_LDS_B128, overlapped with step t's WMMA work.
__global__ __launch_bounds__(1024) void k_lstm(
    const _Float16* __restrict__ whhSwz, const float* __restrict__ xg,
    float* __restrict__ out,
    float* __restrict__ cs_f, float* __restrict__ fs_f,
    float* __restrict__ cs_b, float* __restrict__ fs_b,
    float* __restrict__ h_f, float* __restrict__ h_bflip) {
  extern __shared__ char smem[];
  float*    gAct    = (float*)smem;                    // 16*1024 f32 = 64KB
  _Float16* hBuf    = (_Float16*)(smem + 65536);       // 16*256 f16 =  8KB
  float*    cBuf    = (float*)(smem + 73728);          // 16*256 f32 = 16KB
  float*    xgStage = (float*)(smem + 90112);          // 2 x 16*1024 f32 = 128KB
  const int dir   = blockIdx.x >> 1;
  const int mhalf = blockIdx.x & 1;   // which 16 batches this WGP owns
  const int tid = threadIdx.x, wave = tid >> 5, lane = tid & 31;
  const int mloc = lane & 15, half = lane >> 4;
  const _Float16* wd =
      whhSwz + (size_t)dir * kNTiles * kKTilesHH * kFragHalves;
  const float* xgd = xg + (size_t)dir * kT * kB * kG;
  float* csO = dir ? cs_b : cs_f;
  float* fsO = dir ? fs_b : fs_f;
  float* hO  = dir ? h_bflip : h_f;

  // W_hh fragments resident in VGPRs for the whole time loop.
  v16h Bf0[kKTilesHH], Bf1[kKTilesHH];
  {
    int n0 = wave * 2, n1 = n0 + 1;
#pragma unroll
    for (int k = 0; k < kKTilesHH; ++k) {
      Bf0[k] = *(const v16h*)(wd + (size_t)(n0 * kKTilesHH + k) * kFragHalves + lane * 16);
      Bf1[k] = *(const v16h*)(wd + (size_t)(n1 * kKTilesHH + k) * kFragHalves + lane * 16);
    }
  }
  // Stage xg[t=0] while we zero-init the recurrent state.
  {
    const float* src = xgd + (size_t)(mhalf * 16) * kG;
#pragma unroll
    for (int q = 0; q < 4; ++q)
      async_copy_b128(src + q * 4096 + tid * 4, xgStage + q * 4096 + tid * 4);
  }
  for (int i = tid; i < 16 * kH; i += 1024) {
    cBuf[i] = 0.0f;
    hBuf[i] = (_Float16)0.0f;
  }
  wait_async0();
  __syncthreads();

  const int col0 = wave * 32 + mloc;   // this wave's gate columns
  const int col1 = col0 + 16;
  const int gate = wave >> 3;          // 0:i 1:f 2:g(tanh) 3:o

  for (int t = 0; t < kT; ++t) {
    // Kick off async staging of xg[t+1] into the other LDS buffer.
    if (t + 1 < kT) {
      const float* src = xgd + ((size_t)(t + 1) * kB + mhalf * 16) * kG;
      float* dst = xgStage + ((t + 1) & 1) * 16384;
#pragma unroll
      for (int q = 0; q < 4; ++q)
        async_copy_b128(src + q * 4096 + tid * 4, dst + q * 4096 + tid * 4);
    }
    // Warm L2 two steps ahead so the async engine hits L2, not HBM.
    if (t + 2 < kT)
      __builtin_prefetch(xgd + ((size_t)(t + 2) * kB + mhalf * 16) * kG + tid * 16, 0, 1);

    // g[:, cols] = h_prev @ W_hh[:, cols]
    v8f acc0 = {}, acc1 = {};
#pragma unroll
    for (int k = 0; k < kKTilesHH; ++k) {
      int base = k * 32 + 8 * half;
      union { v16h v; v8h h[2]; } a;
      a.h[0] = *(const v8h*)(hBuf + mloc * kH + base);
      a.h[1] = *(const v8h*)(hBuf + mloc * kH + base + 16);
      acc0 = __builtin_amdgcn_wmma_f32_16x16x32_f16(
          false, a.v, false, Bf0[k], (short)0, acc0, false, false);
      acc1 = __builtin_amdgcn_wmma_f32_16x16x32_f16(
          false, a.v, false, Bf1[k], (short)0, acc1, false, false);
    }
    // add staged input projection (from LDS), activate, publish to LDS
    const float* xs = xgStage + (t & 1) * 16384;
#pragma unroll
    for (int v = 0; v < 8; ++v) {
      int b = v + 8 * half;
      float g0 = acc0[v] + xs[b * kG + col0];
      float g1 = acc1[v] + xs[b * kG + col1];
      gAct[b * kG + col0] = (gate == 2) ? dev_tanh(g0) : dev_sigmoid(g0);
      gAct[b * kG + col1] = (gate == 2) ? dev_tanh(g1) : dev_sigmoid(g1);
    }
    // Drain our async ops (t+1 staging) before the barrier so every wave's
    // staged data is LDS-visible to all waves next iteration.
    wait_async0();
    __syncthreads();
    // elementwise cell update: 16 batches x 256 cols, 4 per thread
    {
      int bl = tid >> 6;
      int j0 = (tid & 63) * 4;
      int bg = mhalf * 16 + bl;
      int th = dir ? (kT - 1 - t) : t;
      const float* gr = gAct + bl * kG;
#pragma unroll
      for (int jj = 0; jj < 4; ++jj) {
        int j = j0 + jj;
        float iv = gr[j], fv = gr[kH + j], gv = gr[2 * kH + j], ov = gr[3 * kH + j];
        float cp = cBuf[bl * kH + j];
        float c  = fv * cp + iv * gv;
        float h  = ov * dev_tanh(c);
        cBuf[bl * kH + j] = c;
        hBuf[bl * kH + j] = (_Float16)h;
        size_t o = ((size_t)bg * kT + t) * kH + j;
        __builtin_nontemporal_store(c, &csO[o]);
        __builtin_nontemporal_store(fv, &fsO[o]);
        __builtin_nontemporal_store(h, &hO[((size_t)bg * kT + th) * kH + j]);
        if (dir == 0 && t == kT - 1) out[bg * 2 * kH + j] = h;
        if (dir == 1 && t == 0)      out[bg * 2 * kH + kH + j] = h;
      }
    }
    __syncthreads();
  }
}

extern "C" void kernel_launch(void* const* d_in, const int* in_sizes, int n_in,
                              void* d_out, int out_size, void* d_ws, size_t ws_size,
                              hipStream_t stream) {
  const float* x     = (const float*)d_in[0];
  const float* Wf_ih = (const float*)d_in[1];
  const float* Wf_hh = (const float*)d_in[2];
  const float* bf    = (const float*)d_in[3];
  const float* Wb_ih = (const float*)d_in[4];
  const float* Wb_hh = (const float*)d_in[5];
  const float* bb    = (const float*)d_in[6];
  float* out = (float*)d_out;

  // Workspace layout:
  //   [0, 512KB)        : swizzled f16 W_ih (both dirs)
  //   [512KB, 1.5MB)    : swizzled f16 W_hh (both dirs)
  //   [1.5MB, +512MB)   : xg f32 [dir][t*32+b][1024]
  char* ws = (char*)d_ws;
  _Float16* wihS = (_Float16*)ws;
  _Float16* whhS = (_Float16*)(ws + 524288);
  float*    xg   = (float*)(ws + 1572864);

  k_swizzle<<<512, 256, 0, stream>>>(Wf_ih, wihS, kKTilesIH);
  k_swizzle<<<512, 256, 0, stream>>>(Wb_ih, wihS + 131072, kKTilesIH);
  k_swizzle<<<1024, 256, 0, stream>>>(Wf_hh, whhS, kKTilesHH);
  k_swizzle<<<1024, 256, 0, stream>>>(Wb_hh, whhS + 262144, kKTilesHH);

  k_xproj<<<16384, 256, 0, stream>>>(x, wihS, bf, bb, xg);

  size_t S = (size_t)kB * kT * kH;  // 16,777,216
  float* o_out = out;
  float* o_csf = out + 16384;
  float* o_fsf = o_csf + S;
  float* o_csb = o_fsf + S;
  float* o_fsb = o_csb + S;
  float* o_hf  = o_fsb + S;
  float* o_hb  = o_hf + S;

  hipFuncSetAttribute(reinterpret_cast<const void*>(&k_lstm),
                      hipFuncAttributeMaxDynamicSharedMemorySize, 221184);
  k_lstm<<<4, 1024, 221184, stream>>>(whhS, xg, o_out, o_csf, o_fsf,
                                      o_csb, o_fsb, o_hf, o_hb);
}
